// LinearCrossAttention_18811956756702
// MI455X (gfx1250) — matrix operand order repaired and checked by reading, not compiled, not run on hardware
//
#include <hip/hip_runtime.h>
#include <math.h>
#include <stdint.h>

#define HEADS 8
#define DHEAD 64
#define BATCH 8
#define CIN   256
#define NPIX  4096
#define HID   512

#define NSLICE 16
#define SL     256   // m-slice length for ctx partial GEMM
#define VPAD   264   // padded LDS row stride for V slice (16B aligned, bank-spread)

typedef __bf16 bf16_t;
typedef bf16_t bf16x16 __attribute__((ext_vector_type(16)));
typedef bf16_t bf16x8  __attribute__((ext_vector_type(8)));
typedef bf16_t bf16x2  __attribute__((ext_vector_type(2)));
typedef float  f32x8   __attribute__((ext_vector_type(8)));
typedef float  f32x4   __attribute__((ext_vector_type(4)));
typedef int    i32x4   __attribute__((vector_size(16)));

#define LDA 40  // padded LDS row stride (bf16 elems): 80B, keeps 16B alignment

__device__ __forceinline__ bf16x16 cat8(bf16x8 lo, bf16x8 hi) {
  return __builtin_shufflevector(lo, hi, 0, 1, 2, 3, 4, 5, 6, 7, 8, 9, 10, 11,
                                 12, 13, 14, 15);
}

__device__ __forceinline__ f32x8 wmma_bf16(bf16x16 a, bf16x16 b, f32x8 c) {
  return __builtin_amdgcn_wmma_f32_16x16x32_bf16(false, a, false, b, (short)0, c,
                                                 false, false);
}

// --- async global->LDS copy (16B per lane), tracked by ASYNCcnt -------------
__device__ __forceinline__ void async_copy16(const bf16_t* g, bf16_t* l) {
#if __has_builtin(__builtin_amdgcn_global_load_async_to_lds_b128)
  __builtin_amdgcn_global_load_async_to_lds_b128(
      (__attribute__((address_space(1))) i32x4*)g,
      (__attribute__((address_space(3))) i32x4*)(uintptr_t)(uint32_t)(uintptr_t)l,
      0, 0);
#else
  asm volatile("global_load_async_to_lds_b128 %0, %1, off"
               :
               : "v"((uint32_t)(uintptr_t)l), "v"(g)
               : "memory");
#endif
}
__device__ __forceinline__ void wait_async0() {
#if __has_builtin(__builtin_amdgcn_s_wait_asynccnt)
  __builtin_amdgcn_s_wait_asynccnt(0);
#else
  asm volatile("s_wait_asynccnt 0" ::: "memory");
#endif
}

// ---------------------------------------------------------------------------
// Qt[b, n, o] = scale * sum_c Wq[o,c] * X[b,c,n]  (transposed store, M=512)
// ---------------------------------------------------------------------------
__global__ __launch_bounds__(256) void proj_q_kernel(
    const float* __restrict__ W, const float* __restrict__ X,
    bf16_t* __restrict__ Qt, float scale) {
  __shared__ bf16_t As[64 * LDA];
  __shared__ bf16_t Bs[128 * LDA];
  const int b = blockIdx.z, m0 = blockIdx.y * 64, n0 = blockIdx.x * 128;
  const int wv = threadIdx.x >> 5, lane = threadIdx.x & 31;
  const int mw = (wv & 3) * 16, nwl = (wv >> 2) * 64;
  const float* Xb = X + (size_t)b * CIN * NPIX;
  bf16_t* Qtb = Qt + (size_t)b * NPIX * HID;
  const int nl = lane & 15;
  const int kb8 = (lane >> 4) * 8, kb16 = (lane >> 4) * 16;
  const int ar = threadIdx.x >> 2, ak = (threadIdx.x & 3) * 8;    // A fill
  const int kp = (threadIdx.x >> 4) * 2, bn = (threadIdx.x & 15) * 8;  // B fill
  const int arow = mw + (lane & 15);

  f32x8 acc[4] = {};
  for (int k0 = 0; k0 < CIN; k0 += 32) {
    __syncthreads();
#pragma unroll
    for (int i = 0; i < 8; ++i)
      As[ar * LDA + ak + i] = (bf16_t)W[(m0 + ar) * CIN + k0 + ak + i];
    const float* s0 = Xb + (size_t)(k0 + kp) * NPIX + n0 + bn;
    const float* s1 = s0 + NPIX;
#pragma unroll
    for (int i = 0; i < 8; ++i) {
      bf16x2 pr = {(bf16_t)s0[i], (bf16_t)s1[i]};
      *(bf16x2*)&Bs[(bn + i) * LDA + kp] = pr;  // transpose, packed b32
    }
    __syncthreads();
    bf16x16 a = cat8(*(const bf16x8*)&As[arow * LDA + kb16],
                     *(const bf16x8*)&As[arow * LDA + kb16 + 8]);
    bf16x16 bb[4];
#pragma unroll
    for (int t = 0; t < 4; ++t) {
      const int cl = nwl + t * 16 + nl;
      bb[t] = cat8(*(const bf16x8*)&Bs[cl * LDA + kb16],
                   *(const bf16x8*)&Bs[cl * LDA + kb16 + 8]);
    }
#pragma unroll
    for (int t = 0; t < 4; ++t) acc[t] = wmma_bf16(a, bb[t], acc[t]);
  }
  const int mstore = m0 + mw + (lane >> 4) * 8;
#pragma unroll
  for (int t = 0; t < 4; ++t) {
    bf16x8 pk;
#pragma unroll
    for (int r = 0; r < 8; ++r) pk[r] = (bf16_t)(acc[t][r] * scale);
    *(bf16x8*)&Qtb[(size_t)(n0 + nwl + t * 16 + nl) * HID + mstore] = pk;
  }
}

// ---------------------------------------------------------------------------
// K[b,o,m], V[b,o,m] = Wk/Wv @ context  (context read once, shared B tile)
// ---------------------------------------------------------------------------
__global__ __launch_bounds__(256) void proj_kv_kernel(
    const float* __restrict__ Wk, const float* __restrict__ Wv,
    const float* __restrict__ X, bf16_t* __restrict__ Kb,
    bf16_t* __restrict__ Vb) {
  __shared__ bf16_t AsK[64 * LDA];
  __shared__ bf16_t AsV[64 * LDA];
  __shared__ bf16_t Bs[128 * LDA];
  const int b = blockIdx.z, m0 = blockIdx.y * 64, n0 = blockIdx.x * 128;
  const int wv = threadIdx.x >> 5, lane = threadIdx.x & 31;
  const int mw = (wv & 3) * 16, nwl = (wv >> 2) * 64;
  const float* Xb = X + (size_t)b * CIN * NPIX;
  bf16_t* Kbb = Kb + (size_t)b * HID * NPIX;
  bf16_t* Vbb = Vb + (size_t)b * HID * NPIX;
  const int nl = lane & 15;
  const int kb8 = (lane >> 4) * 8, kb16 = (lane >> 4) * 16;
  const int ar = threadIdx.x >> 2, ak = (threadIdx.x & 3) * 8;
  const int kp = (threadIdx.x >> 4) * 2, bn = (threadIdx.x & 15) * 8;
  const int arow = mw + (lane & 15);

  f32x8 accK[4] = {}, accV[4] = {};
  for (int k0 = 0; k0 < CIN; k0 += 32) {
    __syncthreads();
#pragma unroll
    for (int i = 0; i < 8; ++i) {
      AsK[ar * LDA + ak + i] = (bf16_t)Wk[(m0 + ar) * CIN + k0 + ak + i];
      AsV[ar * LDA + ak + i] = (bf16_t)Wv[(m0 + ar) * CIN + k0 + ak + i];
    }
    const float* s0 = Xb + (size_t)(k0 + kp) * NPIX + n0 + bn;
    const float* s1 = s0 + NPIX;
#pragma unroll
    for (int i = 0; i < 8; ++i) {
      bf16x2 pr = {(bf16_t)s0[i], (bf16_t)s1[i]};
      *(bf16x2*)&Bs[(bn + i) * LDA + kp] = pr;
    }
    __syncthreads();
    bf16x16 aK = cat8(*(const bf16x8*)&AsK[arow * LDA + kb16],
                      *(const bf16x8*)&AsK[arow * LDA + kb16 + 8]);
    bf16x16 aV = cat8(*(const bf16x8*)&AsV[arow * LDA + kb16],
                      *(const bf16x8*)&AsV[arow * LDA + kb16 + 8]);
    bf16x16 bb[4];
#pragma unroll
    for (int t = 0; t < 4; ++t) {
      const int cl = nwl + t * 16 + nl;
      bb[t] = cat8(*(const bf16x8*)&Bs[cl * LDA + kb16],
                   *(const bf16x8*)&Bs[cl * LDA + kb16 + 8]);
    }
#pragma unroll
    for (int t = 0; t < 4; ++t) {
      accK[t] = wmma_bf16(aK, bb[t], accK[t]);
      accV[t] = wmma_bf16(aV, bb[t], accV[t]);
    }
  }
  const int mstore = m0 + mw + (lane >> 4) * 8;
#pragma unroll
  for (int t = 0; t < 4; ++t)
#pragma unroll
    for (int r = 0; r < 8; ++r) {
      const size_t idx = (size_t)(mstore + r) * NPIX + n0 + nwl + t * 16 + nl;
      Kbb[idx] = (bf16_t)accK[t][r];
      Vbb[idx] = (bf16_t)accV[t][r];
    }
}

// ---------------------------------------------------------------------------
// Row softmax statistics over K
// ---------------------------------------------------------------------------
__global__ __launch_bounds__(256) void rowstats_kernel(
    const bf16_t* __restrict__ Kb, float* __restrict__ rmax,
    float* __restrict__ rsuminv) {
  const int row = blockIdx.x;
  const bf16_t* p = Kb + (size_t)row * NPIX;
  const int wv = threadIdx.x >> 5, lane = threadIdx.x & 31;

  float m = -1e30f;
  for (int i = threadIdx.x; i < NPIX; i += 256) m = fmaxf(m, (float)p[i]);
#pragma unroll
  for (int o = 16; o; o >>= 1) m = fmaxf(m, __shfl_xor(m, o, 32));
  __shared__ float sm[8];
  if (lane == 0) sm[wv] = m;
  __syncthreads();
  m = sm[0];
#pragma unroll
  for (int i = 1; i < 8; ++i) m = fmaxf(m, sm[i]);

  float s = 0.f;
  for (int i = threadIdx.x; i < NPIX; i += 256) s += __expf((float)p[i] - m);
#pragma unroll
  for (int o = 16; o; o >>= 1) s += __shfl_xor(s, o, 32);
  __shared__ float ss[8];
  if (lane == 0) ss[wv] = s;
  __syncthreads();
  if (threadIdx.x == 0) {
    float t = 0.f;
#pragma unroll
    for (int i = 0; i < 8; ++i) t += ss[i];
    rmax[row] = m;
    rsuminv[row] = 1.f / t;
  }
}

// ---------------------------------------------------------------------------
// Partial ctx GEMM over an m-slice: P[s][b*8+h][e][d] = sum_{m in slice}
// exp(K[d,m]-max_d) * V[e,m].  V slice staged in LDS via async global->LDS.
// grid (NSLICE, HEADS, BATCH)
// ---------------------------------------------------------------------------
__global__ __launch_bounds__(256) void ctx_partial_kernel(
    const bf16_t* __restrict__ Kb, const bf16_t* __restrict__ Vb,
    const float* __restrict__ rmax, float* __restrict__ P) {
  __shared__ bf16_t Vs[64 * VPAD];
  const int s = blockIdx.x, h = blockIdx.y, b = blockIdx.z;
  const int wv = threadIdx.x >> 5, lane = threadIdx.x & 31;
  const int m0 = (wv & 3) * 16;   // d-tile
  const int e0 = (wv >> 2) * 32;  // two e-tiles
  const size_t base = ((size_t)b * HID + h * DHEAD) * NPIX;
  const int srow = b * HID + h * DHEAD;
  const int d = m0 + (lane & 15);
  const float dmax = rmax[srow + d];
  const int nl = lane & 15;
  const int kb8 = (lane >> 4) * 8, kb16 = (lane >> 4) * 16;
  const bf16_t* krow = Kb + base + (size_t)d * NPIX + s * SL;

  // stage V slice (64 rows x SL) into LDS asynchronously: 64 elems/thread
  {
    const int frow = threadIdx.x >> 2;
    const int fseg = (threadIdx.x & 3) * 64;
    const bf16_t* src = Vb + base + (size_t)frow * NPIX + s * SL + fseg;
#pragma unroll
    for (int i = 0; i < 8; ++i)
      async_copy16(src + i * 8, &Vs[frow * VPAD + fseg + i * 8]);
  }
  wait_async0();
  __syncthreads();

  f32x8 acc[2] = {};
#pragma unroll
  for (int k0 = 0; k0 < SL; k0 += 32) {
    __builtin_prefetch(krow + k0 + 256, 0, 0);  // global_prefetch
    bf16x8 klo = *(const bf16x8*)(krow + k0 + kb16);
    bf16x8 khi = *(const bf16x8*)(krow + k0 + kb16 + 8);
    bf16x16 a;
#pragma unroll
    for (int j = 0; j < 8; ++j) {
      a[j] = (bf16_t)__expf((float)klo[j] - dmax);
      a[j + 8] = (bf16_t)__expf((float)khi[j] - dmax);
    }
    bf16x16 bb[2];
#pragma unroll
    for (int t = 0; t < 2; ++t) {
      const bf16_t* vr = &Vs[(e0 + t * 16 + nl) * VPAD + k0 + kb16];
      bb[t] = cat8(*(const bf16x8*)vr, *(const bf16x8*)(vr + 8));
    }
#pragma unroll
    for (int t = 0; t < 2; ++t) acc[t] = wmma_bf16(a, bb[t], acc[t]);
  }
  const int dst = m0 + (lane >> 4) * 8;
  float* Pp = P + ((size_t)s * 64 + (b * HEADS + h)) * DHEAD * DHEAD;
#pragma unroll
  for (int t = 0; t < 2; ++t) {
    const int e = e0 + t * 16 + nl;
    f32x4 lo = __builtin_shufflevector(acc[t], acc[t], 0, 1, 2, 3);
    f32x4 hi = __builtin_shufflevector(acc[t], acc[t], 4, 5, 6, 7);
    *(f32x4*)&Pp[(size_t)e * DHEAD + dst] = lo;       // transposed [e][d]
    *(f32x4*)&Pp[(size_t)e * DHEAD + dst + 4] = hi;
  }
}

// ---------------------------------------------------------------------------
// ctxT[b,h,e,d] = bf16( (sum_s P[s][bh][e][d]) * rsuminv[d] )
// ---------------------------------------------------------------------------
__global__ __launch_bounds__(256) void ctx_reduce_kernel(
    const float* __restrict__ P, const float* __restrict__ rsuminv,
    bf16_t* __restrict__ ctxT) {
  const int bh = blockIdx.x;
  const int srow = (bh >> 3) * HID + (bh & 7) * DHEAD;
  for (int i = threadIdx.x; i < DHEAD * DHEAD; i += 256) {
    const int dd = i & 63;
    float s = 0.f;
#pragma unroll
    for (int sl = 0; sl < NSLICE; ++sl)
      s += P[((size_t)sl * 64 + bh) * DHEAD * DHEAD + i];
    ctxT[(size_t)bh * DHEAD * DHEAD + i] = (bf16_t)(s * rsuminv[srow + dd]);
  }
}

// ---------------------------------------------------------------------------
// Ot[b, n, h*64+e] = sum_d ctxT[e,d] * Qt[b, n, h*64+d]   (M=64, K=64)
// ---------------------------------------------------------------------------
__global__ __launch_bounds__(256) void out_kernel(
    const bf16_t* __restrict__ ctxT, const bf16_t* __restrict__ Qt,
    bf16_t* __restrict__ Ot) {
  const int h = blockIdx.y, b = blockIdx.z;
  const int n0 = blockIdx.x * 128;
  const int wv = threadIdx.x >> 5, lane = threadIdx.x & 31;
  const int e0 = (wv & 3) * 16;
  const int nw = n0 + (wv >> 2) * 64;
  const bf16_t* cT = ctxT + (size_t)(b * HEADS + h) * DHEAD * DHEAD;
  const bf16_t* Qtb = Qt + (size_t)b * NPIX * HID + h * DHEAD;
  bf16_t* Otb = Ot + (size_t)b * NPIX * HID + h * DHEAD;
  const int nl = lane & 15;
  const int erow = e0 + (lane & 15);
  const int kb8 = (lane >> 4) * 8, kb16 = (lane >> 4) * 16;

  f32x8 acc[4] = {};
#pragma unroll
  for (int k0 = 0; k0 < DHEAD; k0 += 32) {
    bf16x16 a = cat8(*(const bf16x8*)&cT[erow * DHEAD + k0 + kb16],
                     *(const bf16x8*)&cT[erow * DHEAD + k0 + kb16 + 8]);
    bf16x16 bb[4];
#pragma unroll
    for (int t = 0; t < 4; ++t) {
      const bf16_t* q = Qtb + (size_t)(nw + t * 16 + nl) * HID + k0 + kb16;
      bb[t] = cat8(*(const bf16x8*)q, *(const bf16x8*)(q + 8));
    }
#pragma unroll
    for (int t = 0; t < 4; ++t) acc[t] = wmma_bf16(a, bb[t], acc[t]);
  }
  const int estore = e0 + (lane >> 4) * 8;
#pragma unroll
  for (int t = 0; t < 4; ++t) {
    bf16x8 pk;
#pragma unroll
    for (int r = 0; r < 8; ++r) pk[r] = (bf16_t)acc[t][r];
    *(bf16x8*)&Otb[(size_t)(nw + t * 16 + nl) * HID + estore] = pk;
  }
}

// ---------------------------------------------------------------------------
// Y[b,o,n] = sum_c Wo[o,c] * Ot[b,n,c] + bo[o]   (M=256, K=512, N=4096)
// ---------------------------------------------------------------------------
__global__ __launch_bounds__(256) void final_kernel(
    const float* __restrict__ Wo, const float* __restrict__ bo,
    const bf16_t* __restrict__ Ot, float* __restrict__ Y) {
  __shared__ bf16_t As[64 * LDA];
  const int b = blockIdx.z;
  const int m0 = blockIdx.y * 64;
  const int n0 = blockIdx.x * 128;
  const int wv = threadIdx.x >> 5, lane = threadIdx.x & 31;
  const int mw = (wv & 3) * 16;
  const int nw = n0 + (wv >> 2) * 64;
  const bf16_t* Otb = Ot + (size_t)b * NPIX * HID;
  float* Yb = Y + (size_t)b * CIN * NPIX;
  const int nl = lane & 15;
  const int kb8 = (lane >> 4) * 8, kb16 = (lane >> 4) * 16;
  const int ar = threadIdx.x >> 2, ak = (threadIdx.x & 3) * 8;
  const int arow = mw + (lane & 15);

  f32x8 acc[4] = {};
  for (int k0 = 0; k0 < HID; k0 += 32) {
    __syncthreads();
#pragma unroll
    for (int i = 0; i < 8; ++i)
      As[ar * LDA + ak + i] = (bf16_t)Wo[(m0 + ar) * HID + k0 + ak + i];
    __syncthreads();
    bf16x16 a = cat8(*(const bf16x8*)&As[arow * LDA + kb16],
                     *(const bf16x8*)&As[arow * LDA + kb16 + 8]);
    bf16x16 bb[4];
#pragma unroll
    for (int t = 0; t < 4; ++t) {
      const bf16_t* o = Otb + (size_t)(nw + t * 16 + nl) * HID + k0 + kb16;
      bb[t] = cat8(*(const bf16x8*)o, *(const bf16x8*)(o + 8));
    }
#pragma unroll
    for (int t = 0; t < 4; ++t) acc[t] = wmma_bf16(a, bb[t], acc[t]);
  }
  const int mstore = m0 + mw + (lane >> 4) * 8;
#pragma unroll
  for (int t = 0; t < 4; ++t)
#pragma unroll
    for (int r = 0; r < 8; ++r)
      Yb[(size_t)(mstore + r) * NPIX + nw + t * 16 + nl] =
          acc[t][r] + bo[mstore + r];
}

// ---------------------------------------------------------------------------
extern "C" void kernel_launch(void* const* d_in, const int* in_sizes, int n_in,
                              void* d_out, int out_size, void* d_ws,
                              size_t ws_size, hipStream_t stream) {
  const float* x  = (const float*)d_in[0];
  const float* cx = (const float*)d_in[1];
  const float* Wq = (const float*)d_in[2];
  const float* Wk = (const float*)d_in[3];
  const float* Wv = (const float*)d_in[4];
  const float* Wo = (const float*)d_in[5];
  const float* bo = (const float*)d_in[6];
  float* y = (float*)d_out;

  char* ws = (char*)d_ws;
  const size_t PROJ_BYTES = (size_t)BATCH * HID * NPIX * sizeof(bf16_t); // 32 MiB
  const size_t CTX_BYTES = (size_t)BATCH * HEADS * DHEAD * DHEAD * sizeof(bf16_t);
  bf16_t* Qt = (bf16_t*)(ws);                     // [b][n][hid]
  bf16_t* Kb = (bf16_t*)(ws + PROJ_BYTES);        // [b][hid][m]
  bf16_t* Vb = (bf16_t*)(ws + 2 * PROJ_BYTES);    // [b][hid][m]
  bf16_t* ctxT = (bf16_t*)(ws + 3 * PROJ_BYTES);  // [bh][e][d], 512 KiB
  float* rmax = (float*)(ws + 3 * PROJ_BYTES + CTX_BYTES);
  float* rsuminv = rmax + BATCH * HID;
  float* P = (float*)(ws + 3 * PROJ_BYTES + CTX_BYTES + 2 * 16384); // 16 MiB
  bf16_t* Ot = Kb;  // K buffer dead after ctx partials; reuse as [b][n][hid]

  dim3 blk(256);
  const float scale = 0.125f;  // 64^-0.5

  proj_q_kernel<<<dim3(32, 8, 8), blk, 0, stream>>>(Wq, x, Qt, scale);
  proj_kv_kernel<<<dim3(32, 8, 8), blk, 0, stream>>>(Wk, Wv, cx, Kb, Vb);
  rowstats_kernel<<<dim3(BATCH * HID), blk, 0, stream>>>(Kb, rmax, rsuminv);
  ctx_partial_kernel<<<dim3(NSLICE, HEADS, BATCH), blk, 0, stream>>>(Kb, Vb,
                                                                     rmax, P);
  ctx_reduce_kernel<<<dim3(BATCH * HEADS), blk, 0, stream>>>(P, rsuminv, ctxT);
  out_kernel<<<dim3(32, HEADS, BATCH), blk, 0, stream>>>(ctxT, Qt, Ot);
  final_kernel<<<dim3(32, 4, 8), blk, 0, stream>>>(Wo, bo, Ot, y);
}
